// CausalSelfAttention_23965917511797
// MI455X (gfx1250) — matrix operand order repaired and checked by reading, not compiled, not run on hardware
//
#include <hip/hip_runtime.h>

// ---------------------------------------------------------------------------
// Problem constants (match reference)
// ---------------------------------------------------------------------------
#define S_LEN    2048
#define D_MODEL  4096
#define N_HEADS  32
#define N_KV     8
#define HEAD_DIM 128
#define KV_STRIDE (N_KV * HEAD_DIM)   // 1024
#define NEG_INF  (-1e30f)

typedef __attribute__((ext_vector_type(16))) _Float16 v16h;
typedef __attribute__((ext_vector_type(8)))  _Float16 v8h;
typedef __attribute__((ext_vector_type(8)))  float    v8f;

// ---------------------------------------------------------------------------
// Fragment helpers (CDNA5 v_wmma_f32_16x16x32_f16 layouts, ISA 7.12.2)
//   A (16x32 f16): lane m = L%16, half i -> k = (i/8)*16 + (L/16)*8 + i%8
//   B (32x16 f16): lane n = L%16, half i -> k = (i/8)*16 + (L/16)*8 + i%8
//   C (16x16 f32): lane col n = L%16, vgpr v -> row m = v + 8*(L/16)
// ---------------------------------------------------------------------------
__device__ __forceinline__ v8f vzero8f() {
  v8f r;
#pragma unroll
  for (int i = 0; i < 8; ++i) r[i] = 0.0f;
  return r;
}

__device__ __forceinline__ v16h load_afrag(const _Float16* __restrict__ base,
                                           int ld, int m0, int k0, int lane) {
  const int m  = m0 + (lane & 15);
  const int hi = lane >> 4;
  const _Float16* p = base + (size_t)m * ld + k0 + hi * 8;
  v8h lo  = *(const v8h*)(p);
  v8h hi8 = *(const v8h*)(p + 16);
  return __builtin_shufflevector(lo, hi8, 0, 1, 2, 3, 4, 5, 6, 7,
                                          8, 9, 10, 11, 12, 13, 14, 15);
}

__device__ __forceinline__ v16h load_bfrag(const _Float16* __restrict__ base,
                                           int ld, int n0, int k0, int lane) {
  const int n  = n0 + (lane & 15);
  const int hi = lane >> 4;
  const _Float16* p = base + (size_t)n * ld + k0 + hi * 8;
  v8h lo  = *(const v8h*)(p);
  v8h hi8 = *(const v8h*)(p + 16);
  return __builtin_shufflevector(lo, hi8, 0, 1, 2, 3, 4, 5, 6, 7,
                                          8, 9, 10, 11, 12, 13, 14, 15);
}

__device__ __forceinline__ v8f wmma_f16(v16h a, v16h b, v8f c) {
  return __builtin_amdgcn_wmma_f32_16x16x32_f16(
      /*neg_a=*/false, a, /*neg_b=*/false, b,
      /*c_mod=*/(short)0, c, /*reuse_a=*/false, /*reuse_b=*/false);
}

// ---------------------------------------------------------------------------
// f32 -> f16 conversion
// ---------------------------------------------------------------------------
__global__ void cvt_f32_to_f16(const float* __restrict__ in,
                               _Float16* __restrict__ out, int n) {
  int i = blockIdx.x * blockDim.x + threadIdx.x;
  if (i < n) out[i] = (_Float16)in[i];
}

// ---------------------------------------------------------------------------
// GEMM: C[M,N] = X[M,Kd] * W[N,Kd]^T   (torch Linear), f16 in, f32 accum.
// Each wave owns a 16x64 tile; 4 waves / block -> block tile 16x256.
// grid = (M/16, N/256), block = 128.
// ---------------------------------------------------------------------------
#define ST_F16_ROW  0   // out f16, row-major [M,N]
#define ST_F16_COLT 1   // out f16, transposed [N,M]  (for V^T)
#define ST_F32_ROW  2   // out f32, row-major [M,N]   (final output)

template <int MODE>
__global__ __launch_bounds__(128, 1)
void gemm_nt_wmma(const _Float16* __restrict__ X,
                  const _Float16* __restrict__ W,
                  void* __restrict__ Out,
                  int M, int N, int Kd) {
  const int lane = threadIdx.x & 31;
  const int wave = threadIdx.x >> 5;
  const int m0   = blockIdx.x * 16;
  const int n0   = blockIdx.y * 256 + wave * 64;
  const int hi   = lane >> 4;
  const int nn   = lane & 15;

  v8f acc0 = vzero8f(), acc1 = vzero8f(), acc2 = vzero8f(), acc3 = vzero8f();

  for (int k0 = 0; k0 < Kd; k0 += 32) {
    // warm L2/L0 for the next k-slab of the streamed operand
    if (k0 + 32 < Kd) {
      __builtin_prefetch(X + (size_t)(m0 + nn) * Kd + k0 + 32, 0, 0);
      __builtin_prefetch(W + (size_t)(n0 + nn) * Kd + k0 + 32, 0, 0);
    }
    v16h a  = load_afrag(X, Kd, m0, k0, lane);
    v16h b0 = load_bfrag(W, Kd, n0 +  0, k0, lane);
    v16h b1 = load_bfrag(W, Kd, n0 + 16, k0, lane);
    v16h b2 = load_bfrag(W, Kd, n0 + 32, k0, lane);
    v16h b3 = load_bfrag(W, Kd, n0 + 48, k0, lane);
    acc0 = wmma_f16(a, b0, acc0);
    acc1 = wmma_f16(a, b1, acc1);
    acc2 = wmma_f16(a, b2, acc2);
    acc3 = wmma_f16(a, b3, acc3);
  }

#pragma unroll
  for (int t = 0; t < 4; ++t) {
    const int n = n0 + t * 16 + nn;
    const v8f acc = (t == 0) ? acc0 : (t == 1) ? acc1 : (t == 2) ? acc2 : acc3;
#pragma unroll
    for (int v = 0; v < 8; ++v) {
      const int m = m0 + v + 8 * hi;
      const float val = acc[v];
      if (MODE == ST_F16_ROW) {
        ((_Float16*)Out)[(size_t)m * N + n] = (_Float16)val;
      } else if (MODE == ST_F16_COLT) {
        ((_Float16*)Out)[(size_t)n * M + m] = (_Float16)val;
      } else {
        ((float*)Out)[(size_t)m * N + n] = val;
      }
    }
  }
}

// ---------------------------------------------------------------------------
// RMSNorm over rows of length HEAD_DIM (in-place, f16 storage, f32 math).
// One wave per row, 4 rows per block.
// ---------------------------------------------------------------------------
__global__ void rmsnorm_rows(_Float16* __restrict__ t,
                             const float* __restrict__ w, int nrows) {
  const int row  = blockIdx.x * 4 + (threadIdx.x >> 5);
  const int lane = threadIdx.x & 31;
  if (row >= nrows) return;
  _Float16* p = t + (size_t)row * HEAD_DIM;
  float x[4];
  float ss = 0.0f;
#pragma unroll
  for (int i = 0; i < 4; ++i) {
    x[i] = (float)p[lane + 32 * i];
    ss += x[i] * x[i];
  }
#pragma unroll
  for (int m = 16; m >= 1; m >>= 1) ss += __shfl_xor(ss, m, 32);
  const float inv = rsqrtf(ss * (1.0f / HEAD_DIM) + 1e-6f);
#pragma unroll
  for (int i = 0; i < 4; ++i)
    p[lane + 32 * i] = (_Float16)(x[i] * inv * w[lane + 32 * i]);
}

// ---------------------------------------------------------------------------
// Flash attention with ALiBi + causal mask (GQA G=4).
// One wave per (16-row q tile, head); 4 waves / block.
// Q  : f16 [S, H*HD]  row-major (post-RMSNorm)
// K  : f16 [S, HKV*HD] row-major (post-RMSNorm)
// Vt : f16 [HKV*HD, S]  (transposed -> P.V B-frags contiguous along kv)
// AO : f16 [S, H*HD]
// ---------------------------------------------------------------------------
__global__ __launch_bounds__(128, 1)
void attn_fwd_wmma(const _Float16* __restrict__ Q,
                   const _Float16* __restrict__ K,
                   const _Float16* __restrict__ Vt,
                   _Float16* __restrict__ AO) {
  __shared__ __align__(16) _Float16 plds[4][16][40];  // 40 = 32 + pad

  const int lane = threadIdx.x & 31;
  const int wv   = threadIdx.x >> 5;
  const int h    = blockIdx.y;
  const int q0   = (blockIdx.x * 4 + wv) * 16;
  const int kvh  = h >> 2;                       // G = H/HKV = 4
  const int hi   = lane >> 4;
  const int nn   = lane & 15;

  // ALiBi: n_heads=32 -> slope_h = 2^(-0.25*(h+1))
  const float slope    = exp2f(-0.25f * (float)(h + 1));
  const float sm_scale = 0.08838834764831845f;   // 1/sqrt(128)

  // Q fragments for this tile (d split into 4 k-chunks of 32)
  const _Float16* qbase = Q + (size_t)h * HEAD_DIM;
  v16h qa[4];
#pragma unroll
  for (int j = 0; j < 4; ++j) qa[j] = load_afrag(qbase, D_MODEL, q0, j * 32, lane);

  const _Float16* kbase  = K + (size_t)kvh * HEAD_DIM;
  const _Float16* vtbase = Vt + (size_t)(kvh * HEAD_DIM) * S_LEN;

  v8f m_i, l_i;
  v8f oacc[8];
#pragma unroll
  for (int v = 0; v < 8; ++v) { m_i[v] = -3e38f; l_i[v] = 0.0f; }
#pragma unroll
  for (int d = 0; d < 8; ++d) oacc[d] = vzero8f();

  const int nchunks = (q0 + 16 + 31) / 32;       // causal: kv <= q
  for (int c = 0; c < nchunks; ++c) {
    const int kv0 = c * 32;

    // S = Q K^T for 16(q) x 32(kv): two 16x16 C tiles
    v8f s0 = vzero8f(), s1 = vzero8f();
#pragma unroll
    for (int j = 0; j < 4; ++j) {
      v16h kb0 = load_bfrag(kbase, KV_STRIDE, kv0,      j * 32, lane);
      v16h kb1 = load_bfrag(kbase, KV_STRIDE, kv0 + 16, j * 32, lane);
      s0 = wmma_f16(qa[j], kb0, s0);
      s1 = wmma_f16(qa[j], kb1, s1);
    }

    // online softmax (row-wise; row m = v + 8*hi lives in one 16-lane half)
    v8f av, p0, p1;
#pragma unroll
    for (int v = 0; v < 8; ++v) {
      const int m    = v + 8 * hi;
      const int qpos = q0 + m;
      const int kp0  = kv0 + nn;
      const int kp1  = kv0 + 16 + nn;
      const float b0 = (kp0 > qpos) ? NEG_INF : slope * (float)(kp0 - qpos);
      const float b1 = (kp1 > qpos) ? NEG_INF : slope * (float)(kp1 - qpos);
      const float sv0 = s0[v] * sm_scale + b0;
      const float sv1 = s1[v] * sm_scale + b1;

      float rmax = fmaxf(sv0, sv1);
#pragma unroll
      for (int msk = 8; msk >= 1; msk >>= 1)
        rmax = fmaxf(rmax, __shfl_xor(rmax, msk, 16));

      const float mnew  = fmaxf(m_i[v], rmax);
      const float alpha = __expf(m_i[v] - mnew);
      const float e0 = __expf(sv0 - mnew);
      const float e1 = __expf(sv1 - mnew);
      float rsum = e0 + e1;
#pragma unroll
      for (int msk = 8; msk >= 1; msk >>= 1)
        rsum += __shfl_xor(rsum, msk, 16);

      l_i[v] = l_i[v] * alpha + rsum;
      m_i[v] = mnew;
      av[v] = alpha;
      p0[v] = e0;
      p1[v] = e1;
    }

    // rescale running output with whole-vector multiplies (keeps WMMA
    // accumulators in place; no per-element insert/extract churn)
#pragma unroll
    for (int d = 0; d < 8; ++d) oacc[d] *= av;

    // C-frag layout -> A-frag layout through LDS (same-wave LDS is in-order)
#pragma unroll
    for (int v = 0; v < 8; ++v) {
      const int m = v + 8 * hi;
      plds[wv][m][nn]      = (_Float16)p0[v];
      plds[wv][m][16 + nn] = (_Float16)p1[v];
    }
    __builtin_amdgcn_wave_barrier();
    v16h pa;
    {
      const _Float16* pr = &plds[wv][nn][0];
      v8h lo  = *(const v8h*)(pr + hi * 8);
      v8h hi8 = *(const v8h*)(pr + 16 + hi * 8);
      pa = __builtin_shufflevector(lo, hi8, 0, 1, 2, 3, 4, 5, 6, 7,
                                            8, 9, 10, 11, 12, 13, 14, 15);
    }
    __builtin_amdgcn_wave_barrier();

    // O += P V : B-frags from Vt, contiguous along kv
#pragma unroll
    for (int d = 0; d < 8; ++d) {
      v16h vb = load_bfrag(vtbase, S_LEN, d * 16, kv0, lane);
      oacc[d] = wmma_f16(pa, vb, oacc[d]);
    }
  }

  // epilogue: normalize (vector-wise) and store f16
  v8f invl;
#pragma unroll
  for (int v = 0; v < 8; ++v) invl[v] = 1.0f / l_i[v];
#pragma unroll
  for (int d = 0; d < 8; ++d) {
    const v8f o = oacc[d] * invl;
    const int col = h * HEAD_DIM + d * 16 + nn;
#pragma unroll
    for (int v = 0; v < 8; ++v) {
      const int m = q0 + v + 8 * hi;
      AO[(size_t)m * D_MODEL + col] = (_Float16)o[v];
    }
  }
}

// ---------------------------------------------------------------------------
// Launcher
// ---------------------------------------------------------------------------
extern "C" void kernel_launch(void* const* d_in, const int* in_sizes, int n_in,
                              void* d_out, int out_size, void* d_ws, size_t ws_size,
                              hipStream_t stream) {
  (void)in_sizes; (void)n_in; (void)out_size; (void)ws_size;

  const float* x   = (const float*)d_in[0];   // [S, DM]
  const float* Wq  = (const float*)d_in[1];   // [4096, DM]
  const float* Wk  = (const float*)d_in[2];   // [1024, DM]
  const float* Wv  = (const float*)d_in[3];   // [1024, DM]
  const float* Wo  = (const float*)d_in[4];   // [DM, 4096] (already .T'd)
  const float* qnw = (const float*)d_in[5];   // [128]
  const float* knw = (const float*)d_in[6];   // [128]

  char* ws = (char*)d_ws;
  _Float16* xh  = (_Float16*)ws; ws += (size_t)S_LEN * D_MODEL * 2;
  _Float16* wqh = (_Float16*)ws; ws += (size_t)D_MODEL * D_MODEL * 2;
  _Float16* wkh = (_Float16*)ws; ws += (size_t)KV_STRIDE * D_MODEL * 2;
  _Float16* wvh = (_Float16*)ws; ws += (size_t)KV_STRIDE * D_MODEL * 2;
  _Float16* woh = (_Float16*)ws; ws += (size_t)D_MODEL * D_MODEL * 2;
  _Float16* qb  = (_Float16*)ws; ws += (size_t)S_LEN * D_MODEL * 2;
  _Float16* kb  = (_Float16*)ws; ws += (size_t)S_LEN * KV_STRIDE * 2;
  _Float16* vtb = (_Float16*)ws; ws += (size_t)KV_STRIDE * S_LEN * 2;
  _Float16* ao  = (_Float16*)ws; ws += (size_t)S_LEN * D_MODEL * 2;

  const int nx  = S_LEN * D_MODEL;
  const int nwq = D_MODEL * D_MODEL;
  const int nwk = KV_STRIDE * D_MODEL;

  // 1) convert everything to f16
  cvt_f32_to_f16<<<(nx  + 255) / 256, 256, 0, stream>>>(x,  xh,  nx);
  cvt_f32_to_f16<<<(nwq + 255) / 256, 256, 0, stream>>>(Wq, wqh, nwq);
  cvt_f32_to_f16<<<(nwk + 255) / 256, 256, 0, stream>>>(Wk, wkh, nwk);
  cvt_f32_to_f16<<<(nwk + 255) / 256, 256, 0, stream>>>(Wv, wvh, nwk);
  cvt_f32_to_f16<<<(nwq + 255) / 256, 256, 0, stream>>>(Wo, woh, nwq);

  // 2) projections (WMMA)
  {
    dim3 gq(S_LEN / 16, D_MODEL / 256);
    gemm_nt_wmma<ST_F16_ROW><<<gq, 128, 0, stream>>>(xh, wqh, qb, S_LEN, D_MODEL, D_MODEL);
    dim3 gk(S_LEN / 16, KV_STRIDE / 256);
    gemm_nt_wmma<ST_F16_ROW><<<gk, 128, 0, stream>>>(xh, wkh, kb, S_LEN, KV_STRIDE, D_MODEL);
    gemm_nt_wmma<ST_F16_COLT><<<gk, 128, 0, stream>>>(xh, wvh, vtb, S_LEN, KV_STRIDE, D_MODEL);
  }

  // 3) RMSNorm on q and k head-rows
  {
    const int qrows = S_LEN * N_HEADS;
    const int krows = S_LEN * N_KV;
    rmsnorm_rows<<<(qrows + 3) / 4, 128, 0, stream>>>(qb, qnw, qrows);
    rmsnorm_rows<<<(krows + 3) / 4, 128, 0, stream>>>(kb, knw, krows);
  }

  // 4) flash attention
  {
    dim3 ga(S_LEN / 64, N_HEADS);
    attn_fwd_wmma<<<ga, 128, 0, stream>>>(qb, kb, vtb, ao);
  }

  // 5) output projection -> f32 d_out
  {
    dim3 go(S_LEN / 16, D_MODEL / 256);
    gemm_nt_wmma<ST_F32_ROW><<<go, 128, 0, stream>>>(ao, woh, (float*)d_out,
                                                     S_LEN, D_MODEL, D_MODEL);
  }
}
